// GraphTransformer_63952063038066
// MI455X (gfx1250) — matrix operand order repaired and checked by reading, not compiled, not run on hardware
//
#include <hip/hip_runtime.h>
#include <hip/hip_bf16.h>
#include <math.h>

// GraphTransformer on gfx1250: WMMA bf16 GEMMs + VALU edge phase with hw f32 atomics.

typedef __attribute__((ext_vector_type(16))) __bf16 v16bf;
typedef __attribute__((ext_vector_type(8)))  __bf16 v8bf;
typedef __attribute__((ext_vector_type(8)))  float  v8f;

// ---------------------------------------------------------------------------
// WMMA helpers (layouts per CDNA5 ISA 7.12.2, wave32)
// ---------------------------------------------------------------------------
__device__ __forceinline__ v8f wmma_bf16(v16bf a, v16bf b, v8f c) {
  return __builtin_amdgcn_wmma_f32_16x16x32_bf16(
      /*neg_a=*/false, a, /*neg_b=*/false, b,
      /*c_mod=*/(short)0, c, /*reuse_a=*/false, /*reuse_b=*/false);
}

// A fragment (16x32 bf16) from row-major f32 tile: lane<16 row M=lane,
// K = kbase+{0..7,16..23}; lane>=16 row M=lane-16, K = kbase+{8..15,24..31}.
__device__ __forceinline__ v16bf load_A_f32(const float* rowptr, int kbase, int lane) {
  int k0 = kbase + ((lane & 16) ? 8 : 0);
  const float* p = rowptr + k0;
  float4 f0 = *(const float4*)(p);
  float4 f1 = *(const float4*)(p + 4);
  float4 f2 = *(const float4*)(p + 16);
  float4 f3 = *(const float4*)(p + 20);
  v16bf a;
  a[0]=(__bf16)f0.x; a[1]=(__bf16)f0.y; a[2]=(__bf16)f0.z; a[3]=(__bf16)f0.w;
  a[4]=(__bf16)f1.x; a[5]=(__bf16)f1.y; a[6]=(__bf16)f1.z; a[7]=(__bf16)f1.w;
  a[8]=(__bf16)f2.x; a[9]=(__bf16)f2.y; a[10]=(__bf16)f2.z; a[11]=(__bf16)f2.w;
  a[12]=(__bf16)f3.x; a[13]=(__bf16)f3.y; a[14]=(__bf16)f3.z; a[15]=(__bf16)f3.w;
  return a;
}

// Pre-swizzled B fragment: frag f = kt*4+nt; 256 dwords per frag, 8 per lane.
__device__ __forceinline__ v16bf load_B(const unsigned int* frags, int mat, int kt, int nt, int lane) {
  const unsigned int* p = frags + (size_t)mat * 2048 + (kt * 4 + nt) * 256 + lane * 8;
  return *(const v16bf*)p;
}

// C tile (nt-th 16x16): VGPR r -> row rbase+r, col = nt*16 + (lane&15).
__device__ __forceinline__ void c_to_lds(__bf16* tile, const v8f* c, int lane) {
  int col = lane & 15;
  int rbase = (lane & 16) ? 8 : 0;
#pragma unroll
  for (int nt = 0; nt < 4; ++nt)
#pragma unroll
    for (int r = 0; r < 8; ++r)
      tile[(rbase + r) * 64 + nt * 16 + col] = (__bf16)c[nt][r];
}

__device__ __forceinline__ v16bf a_from_lds(const __bf16* tile, int kbase, int lane) {
  int m = lane & 15;
  int k0 = kbase + ((lane & 16) ? 8 : 0);
  v8bf lo = *(const v8bf*)(tile + m * 64 + k0);
  v8bf hi = *(const v8bf*)(tile + m * 64 + k0 + 16);
  return __builtin_shufflevector(lo, hi, 0,1,2,3,4,5,6,7,8,9,10,11,12,13,14,15);
}

__device__ __forceinline__ void store_c_global(float* out, int row0, const v8f* c, int lane, int N) {
  int col = lane & 15;
  int rbase = (lane & 16) ? 8 : 0;
#pragma unroll
  for (int nt = 0; nt < 4; ++nt)
#pragma unroll
    for (int r = 0; r < 8; ++r) {
      int row = row0 + rbase + r;
      if (row < N) out[(size_t)row * 64 + nt * 16 + col] = c[nt][r];
    }
}

// One [16 rows] x [64,64] GEMM: 8 WMMAs, bias, optional relu -> c[4] tiles.
__device__ __forceinline__ void gemm4(v16bf a0, v16bf a1, const unsigned int* frags, int mat,
                                      const float* bias, bool relu, v8f* c, int lane) {
#pragma unroll
  for (int nt = 0; nt < 4; ++nt) {
    v16bf b0 = load_B(frags, mat, 0, nt, lane);
    v16bf b1 = load_B(frags, mat, 1, nt, lane);
    v8f acc = {};
    acc = wmma_bf16(a0, b0, acc);
    acc = wmma_bf16(a1, b1, acc);
    float bv = bias[nt * 16 + (lane & 15)];
#pragma unroll
    for (int r = 0; r < 8; ++r) {
      float t = acc[r] + bv;
      acc[r] = relu ? fmaxf(t, 0.0f) : t;
    }
    c[nt] = acc;
  }
}

// ---------------------------------------------------------------------------
// Weight pre-swizzle: f32 [64,64] row-major -> bf16 WMMA B-fragments.
// ---------------------------------------------------------------------------
struct PrepArgs { const float* W[17]; };

__global__ __launch_bounds__(256) void prep_weights(PrepArgs args, unsigned int* frags) {
  int mat = blockIdx.x;
  const float* W = args.W[mat];
  int f = threadIdx.x >> 5;      // fragment 0..7
  int lane = threadIdx.x & 31;
  int kt = f >> 2, nt = f & 3;
  int n = nt * 16 + (lane & 15);
  int kstart = kt * 32 + ((lane & 16) ? 8 : 0);
  unsigned int* dst = frags + (size_t)mat * 2048 + f * 256 + lane * 8;
#pragma unroll
  for (int j = 0; j < 8; ++j) {
    __bf16 blo = (__bf16)W[(kstart + 2 * j + (j >= 4 ? 8 : 0)) * 64 + n];
    __bf16 bhi = (__bf16)W[(kstart + 2 * j + 1 + (j >= 4 ? 8 : 0)) * 64 + n];
    unsigned int lo = __builtin_bit_cast(unsigned short, blo);
    unsigned int hi = __builtin_bit_cast(unsigned short, bhi);
    dst[j] = lo | (hi << 16);
  }
}

// ---------------------------------------------------------------------------
// Edge embedding MLP (6->16->16->16, relu x2) + mask, one thread per edge.
// ---------------------------------------------------------------------------
__global__ __launch_bounds__(256) void edge_mlp_kernel(
    const float* __restrict__ eattr,
    const float* __restrict__ W1, const float* __restrict__ c1,
    const float* __restrict__ W2, const float* __restrict__ c2,
    const float* __restrict__ W3, const float* __restrict__ c3,
    float* __restrict__ emb, float* __restrict__ mask, int E) {
  __shared__ float sW1[96], sc1[16], sW2[256], sc2[16], sW3[256], sc3[16];
  int tx = threadIdx.x;
  if (tx < 96) sW1[tx] = W1[tx];
  if (tx < 16) { sc1[tx] = c1[tx]; sc2[tx] = c2[tx]; sc3[tx] = c3[tx]; }
  sW2[tx] = W2[tx];
  sW3[tx] = W3[tx];
  __syncthreads();
  int e = blockIdx.x * 256 + tx;
  if (e >= E) return;
  const float4* ap = (const float4*)(eattr + (size_t)e * 8);
  float4 a0 = ap[0], a1 = ap[1];
  float in6[6] = {a0.x, a0.y, a0.z, a0.w, a1.x, a1.y};
  mask[e] = a1.z * (1.0f - a1.w);
  float h1[16], h2[16];
#pragma unroll
  for (int n = 0; n < 16; ++n) {
    float acc = sc1[n];
#pragma unroll
    for (int j = 0; j < 6; ++j) acc = fmaf(in6[j], sW1[j * 16 + n], acc);
    h1[n] = fmaxf(acc, 0.0f);
  }
#pragma unroll
  for (int n = 0; n < 16; ++n) {
    float acc = sc2[n];
#pragma unroll
    for (int j = 0; j < 16; ++j) acc = fmaf(h1[j], sW2[j * 16 + n], acc);
    h2[n] = fmaxf(acc, 0.0f);
  }
  float o[16];
#pragma unroll
  for (int n = 0; n < 16; ++n) {
    float acc = sc3[n];
#pragma unroll
    for (int j = 0; j < 16; ++j) acc = fmaf(h2[j], sW3[j * 16 + n], acc);
    o[n] = acc;
  }
  float4* op = (float4*)(emb + (size_t)e * 16);
  op[0] = make_float4(o[0], o[1], o[2], o[3]);
  op[1] = make_float4(o[4], o[5], o[6], o[7]);
  op[2] = make_float4(o[8], o[9], o[10], o[11]);
  op[3] = make_float4(o[12], o[13], o[14], o[15]);
}

// ---------------------------------------------------------------------------
// Fused input MLP: h = L2(relu(L1(relu(L0(x))))), 16 rows/wave, WMMA.
// ---------------------------------------------------------------------------
__global__ __launch_bounds__(256) void node_mlp_kernel(
    const float* __restrict__ x, const unsigned int* __restrict__ frags,
    const float* __restrict__ b0, const float* __restrict__ b1, const float* __restrict__ b2,
    float* __restrict__ h, int N) {
  __shared__ __bf16 tiles[8][1024];
  int wave = threadIdx.x >> 5, lane = threadIdx.x & 31;
  int row0 = (blockIdx.x * 8 + wave) * 16;
  int rowm = row0 + (lane & 15);
  if (rowm > N - 1) rowm = N - 1;
  const float* rp = x + (size_t)rowm * 64;
  v16bf a0 = load_A_f32(rp, 0, lane);
  v16bf a1 = load_A_f32(rp, 32, lane);
  v8f c[4];
  gemm4(a0, a1, frags, 0, b0, true, c, lane);
  c_to_lds(tiles[wave], c, lane);
  __syncthreads();
  a0 = a_from_lds(tiles[wave], 0, lane);
  a1 = a_from_lds(tiles[wave], 32, lane);
  __syncthreads();
  gemm4(a0, a1, frags, 1, b1, true, c, lane);
  c_to_lds(tiles[wave], c, lane);
  __syncthreads();
  a0 = a_from_lds(tiles[wave], 0, lane);
  a1 = a_from_lds(tiles[wave], 32, lane);
  gemm4(a0, a1, frags, 2, b2, false, c, lane);
  store_c_global(h, row0, c, lane, N);
}

// ---------------------------------------------------------------------------
// Per conv layer: q/k/v projections + skip (skip seeds the scatter accumulator).
// ---------------------------------------------------------------------------
__global__ __launch_bounds__(256) void qkv_kernel(
    const float* __restrict__ h, const unsigned int* __restrict__ frags, int matBase,
    const float* __restrict__ bq, const float* __restrict__ bk,
    const float* __restrict__ bv, const float* __restrict__ bs,
    float* __restrict__ q, float* __restrict__ k, float* __restrict__ v,
    float* __restrict__ agg, int N) {
  int wave = threadIdx.x >> 5, lane = threadIdx.x & 31;
  int row0 = (blockIdx.x * 8 + wave) * 16;
  int rowm = row0 + (lane & 15);
  if (rowm > N - 1) rowm = N - 1;
  const float* rp = h + (size_t)rowm * 64;
  v16bf a0 = load_A_f32(rp, 0, lane);
  v16bf a1 = load_A_f32(rp, 32, lane);
  v8f c[4];
  gemm4(a0, a1, frags, matBase + 0, bq, false, c, lane); store_c_global(q, row0, c, lane, N);
  gemm4(a0, a1, frags, matBase + 1, bk, false, c, lane); store_c_global(k, row0, c, lane, N);
  gemm4(a0, a1, frags, matBase + 2, bv, false, c, lane); store_c_global(v, row0, c, lane, N);
  gemm4(a0, a1, frags, matBase + 3, bs, false, c, lane); store_c_global(agg, row0, c, lane, N);
}

// ---------------------------------------------------------------------------
// Edge attention: 16 lanes/edge, 4 channels/lane. evec = emb@We (LDS weights),
// alpha = dot(q_i, k_j+e)/4 * mask (shfl reduce over 4-lane head groups),
// scatter agg[dst] += (v_j+e)*alpha with hardware f32 atomics (L2-resident).
// ---------------------------------------------------------------------------
__global__ __launch_bounds__(256) void edge_attn_kernel(
    const int* __restrict__ ei, const float* __restrict__ emb, const float* __restrict__ mask,
    const float* __restrict__ We, const float* __restrict__ q, const float* __restrict__ k,
    const float* __restrict__ v, float* __restrict__ agg, int E) {
  __shared__ float sWe[1024];
  for (int i = threadIdx.x; i < 1024; i += 256) sWe[i] = We[i];
  __syncthreads();
  int e = blockIdx.x * 16 + (threadIdx.x >> 4);
  if (e >= E) return;
  int t = threadIdx.x & 15;
  int src = ei[e];
  int dst = ei[E + e];
  float mval = mask[e];
  const float4* ep = (const float4*)(emb + (size_t)e * 16);
  float4 e0 = ep[0], e1 = ep[1], e2 = ep[2], e3 = ep[3];
  float em[16] = {e0.x, e0.y, e0.z, e0.w, e1.x, e1.y, e1.z, e1.w,
                  e2.x, e2.y, e2.z, e2.w, e3.x, e3.y, e3.z, e3.w};
  int c0 = t * 4;
  float evx = 0.f, evy = 0.f, evz = 0.f, evw = 0.f;
#pragma unroll
  for (int j = 0; j < 16; ++j) {
    float4 w = *(const float4*)(sWe + j * 64 + c0);
    evx = fmaf(em[j], w.x, evx);
    evy = fmaf(em[j], w.y, evy);
    evz = fmaf(em[j], w.z, evz);
    evw = fmaf(em[j], w.w, evw);
  }
  float4 qi = *(const float4*)(q + (size_t)dst * 64 + c0);
  float4 kj = *(const float4*)(k + (size_t)src * 64 + c0);
  float4 vj = *(const float4*)(v + (size_t)src * 64 + c0);
  kj.x += evx; kj.y += evy; kj.z += evz; kj.w += evw;
  vj.x += evx; vj.y += evy; vj.z += evz; vj.w += evw;
  float part = qi.x * kj.x + qi.y * kj.y + qi.z * kj.z + qi.w * kj.w;
  part += __shfl_xor(part, 1);
  part += __shfl_xor(part, 2);           // per-head dot over 16 channels
  float alpha = part * 0.25f * mval;     // 1/sqrt(HID) = 0.25
  float* ap = agg + (size_t)dst * 64 + c0;
  unsafeAtomicAdd(ap + 0, vj.x * alpha);
  unsafeAtomicAdd(ap + 1, vj.y * alpha);
  unsafeAtomicAdd(ap + 2, vj.z * alpha);
  unsafeAtomicAdd(ap + 3, vj.w * alpha);
}

// ---------------------------------------------------------------------------
// Output MLP: relu(L0), relu(L1) via WMMA, then per-row dot with w2 + sigmoid.
// ---------------------------------------------------------------------------
__global__ __launch_bounds__(256) void out_mlp_kernel(
    const float* __restrict__ h, const unsigned int* __restrict__ frags,
    const float* __restrict__ b0, const float* __restrict__ b1,
    const float* __restrict__ w2, const float* __restrict__ b2,
    float* __restrict__ out, int N) {
  __shared__ __bf16 tiles[8][1024];
  __shared__ float sw2[64];
  if (threadIdx.x < 64) sw2[threadIdx.x] = w2[threadIdx.x];
  int wave = threadIdx.x >> 5, lane = threadIdx.x & 31;
  int row0 = (blockIdx.x * 8 + wave) * 16;
  int rowm = row0 + (lane & 15);
  if (rowm > N - 1) rowm = N - 1;
  const float* rp = h + (size_t)rowm * 64;
  v16bf a0 = load_A_f32(rp, 0, lane);
  v16bf a1 = load_A_f32(rp, 32, lane);
  v8f c[4];
  gemm4(a0, a1, frags, 15, b0, true, c, lane);
  c_to_lds(tiles[wave], c, lane);
  __syncthreads();
  a0 = a_from_lds(tiles[wave], 0, lane);
  a1 = a_from_lds(tiles[wave], 32, lane);
  __syncthreads();
  gemm4(a0, a1, frags, 16, b1, true, c, lane);
  c_to_lds(tiles[wave], c, lane);
  __syncthreads();
  int row = lane >> 1, half = lane & 1;   // 2 lanes per row
  const __bf16* hr = tiles[wave] + row * 64 + half * 32;
  float acc = 0.f;
#pragma unroll
  for (int j = 0; j < 32; ++j) acc = fmaf((float)hr[j], sw2[half * 32 + j], acc);
  acc += __shfl_xor(acc, 1);
  int grow = row0 + row;
  if (half == 0 && grow < N)
    out[grow] = 1.0f / (1.0f + __expf(-(acc + b2[0])));
}

// ---------------------------------------------------------------------------
extern "C" void kernel_launch(void* const* d_in, const int* in_sizes, int n_in,
                              void* d_out, int out_size, void* d_ws, size_t ws_size,
                              hipStream_t stream) {
  const int N = in_sizes[0] / 64;
  const int E = in_sizes[2] / 8;
  const float* x = (const float*)d_in[0];
  const int* ei = (const int*)d_in[1];
  const float* eattr = (const float*)d_in[2];
  int idx = 3;
  const float *iW[3], *ib[3];
  for (int l = 0; l < 3; ++l) { iW[l] = (const float*)d_in[idx++]; ib[l] = (const float*)d_in[idx++]; }
  const float *eW[3], *ebias[3];
  for (int l = 0; l < 3; ++l) { eW[l] = (const float*)d_in[idx++]; ebias[l] = (const float*)d_in[idx++]; }
  const float *Wq[3], *bq[3], *Wk[3], *bk[3], *Wv[3], *bv[3], *We[3], *Wsk[3], *bsk[3];
  for (int l = 0; l < 3; ++l) {
    Wq[l]  = (const float*)d_in[idx++]; bq[l]  = (const float*)d_in[idx++];
    Wk[l]  = (const float*)d_in[idx++]; bk[l]  = (const float*)d_in[idx++];
    Wv[l]  = (const float*)d_in[idx++]; bv[l]  = (const float*)d_in[idx++];
    We[l]  = (const float*)d_in[idx++];
    Wsk[l] = (const float*)d_in[idx++]; bsk[l] = (const float*)d_in[idx++];
  }
  const float *oW[3], *ob[3];
  for (int l = 0; l < 3; ++l) { oW[l] = (const float*)d_in[idx++]; ob[l] = (const float*)d_in[idx++]; }

  // Workspace layout
  float* ws = (float*)d_ws;
  size_t n64 = (size_t)N * 64;
  float* h    = ws;
  float* q    = h + n64;
  float* kk   = q + n64;
  float* vv   = kk + n64;
  float* agg  = vv + n64;
  float* emb  = agg + n64;
  float* mk   = emb + (size_t)E * 16;
  unsigned int* frags = (unsigned int*)(mk + E);  // 17 matrices * 8KB bf16 fragments

  PrepArgs pa;
  pa.W[0] = iW[0]; pa.W[1] = iW[1]; pa.W[2] = iW[2];
  for (int l = 0; l < 3; ++l) {
    pa.W[3 + 4 * l + 0] = Wq[l];
    pa.W[3 + 4 * l + 1] = Wk[l];
    pa.W[3 + 4 * l + 2] = Wv[l];
    pa.W[3 + 4 * l + 3] = Wsk[l];
  }
  pa.W[15] = oW[0]; pa.W[16] = oW[1];

  prep_weights<<<17, 256, 0, stream>>>(pa, frags);
  edge_mlp_kernel<<<(E + 255) / 256, 256, 0, stream>>>(
      eattr, eW[0], ebias[0], eW[1], ebias[1], eW[2], ebias[2], emb, mk, E);
  int gN = (N + 127) / 128;
  node_mlp_kernel<<<gN, 256, 0, stream>>>(x, frags, ib[0], ib[1], ib[2], h, N);

  float* hin = h;
  float* hout = agg;
  for (int l = 0; l < 3; ++l) {
    qkv_kernel<<<gN, 256, 0, stream>>>(hin, frags, 3 + 4 * l,
                                       bq[l], bk[l], bv[l], bsk[l],
                                       q, kk, vv, hout, N);
    edge_attn_kernel<<<(E + 15) / 16, 256, 0, stream>>>(
        ei, emb, mk, We[l], q, kk, vv, hout, E);
    float* tmp = hin; hin = hout; hout = tmp;
  }
  out_mlp_kernel<<<gN, 256, 0, stream>>>(hin, frags, ob[0], ob[1], oW[2], ob[2],
                                         (float*)d_out, N);
}
